// QwenAttention_19653770346794
// MI455X (gfx1250) — compile-verified, matrix operand order checked by reading, not run on hardware
//
#include <hip/hip_runtime.h>
#include <hip/hip_bf16.h>

typedef __bf16 bf16_t;
typedef __attribute__((ext_vector_type(16))) __bf16 v16bf;
typedef __attribute__((ext_vector_type(8)))  float  v8f;

#define WMMA_BF16(a, b, c) \
  __builtin_amdgcn_wmma_f32_16x16x32_bf16(false, (a), false, (b), (short)0, (c), false, false)

static constexpr int B_   = 2;
static constexpr int S_   = 2048;
static constexpr int H_   = 896;
static constexpr int NH_  = 14;
static constexpr int NKV_ = 2;
static constexpr int HD_  = 64;

// Load a 16-element bf16 fragment as two 16-byte chunks (A/B WMMA layouts are
// built from two contiguous 8-bf16 runs per lane).
__device__ __forceinline__ v16bf load_frag16(const bf16_t* p0, const bf16_t* p1) {
  v16bf r;
  uint4* u = reinterpret_cast<uint4*>(&r);
  u[0] = *reinterpret_cast<const uint4*>(p0);
  u[1] = *reinterpret_cast<const uint4*>(p1);
  return r;
}

// ---------------------------------------------------------------- f32 -> bf16
__global__ void cvt_bf16_kernel(const float* __restrict__ src,
                                bf16_t* __restrict__ dst, int n) {
  int i = blockIdx.x * blockDim.x + threadIdx.x;
  if (i < n) dst[i] = (bf16_t)src[i];
}

// ------------------------------------------------- WMMA GEMM: C = A * W^T + b
// A: [M,K] bf16 row-major, W: [N,K] bf16 row-major (so B[k][n] = W[n][k]).
// Each wave computes a 32(M) x 64(N) tile: 2 A-frags + 4 B-frags -> 8 WMMAs
// per K-step (1.5 b128 loads per WMMA). Block = 8 waves stacked along M
// (256M x 64N block tile) so all waves share the same B stream.
// mode 0: float out row-major [M,N].
// mode 2: bf16 out, V-transposed: (m,n) -> vt[b][kv][d][s]
__global__ void gemm_wmma_kernel(const bf16_t* __restrict__ A,
                                 const bf16_t* __restrict__ W,
                                 const float*  __restrict__ bias,
                                 void* __restrict__ out,
                                 int M, int N, int K, int mode) {
  const int lane = threadIdx.x & 31;
  const int wave = threadIdx.x >> 5;
  const int lr   = lane & 15;
  const int hi   = lane >> 4;      // 0 or 1
  const int d0   = hi * 8;

  const int m0 = blockIdx.x * 256 + wave * 32;
  const int n0 = blockIdx.y * 64;

  const bf16_t* arow0 = A + (size_t)(m0 + lr) * K;
  const bf16_t* arow1 = A + (size_t)(m0 + 16 + lr) * K;
  const bf16_t* wrow  = W + (size_t)(n0 + lr) * K;
  const size_t  wstep = (size_t)16 * K;

  v8f acc[2][4] = {};
  for (int kk = 0; kk < K; kk += 32) {
    v16bf a0 = load_frag16(arow0 + kk + d0, arow0 + kk + d0 + 16);
    v16bf a1 = load_frag16(arow1 + kk + d0, arow1 + kk + d0 + 16);
    v16bf b0 = load_frag16(wrow + 0 * wstep + kk + d0, wrow + 0 * wstep + kk + d0 + 16);
    v16bf b1 = load_frag16(wrow + 1 * wstep + kk + d0, wrow + 1 * wstep + kk + d0 + 16);
    v16bf b2 = load_frag16(wrow + 2 * wstep + kk + d0, wrow + 2 * wstep + kk + d0 + 16);
    v16bf b3 = load_frag16(wrow + 3 * wstep + kk + d0, wrow + 3 * wstep + kk + d0 + 16);
    acc[0][0] = WMMA_BF16(a0, b0, acc[0][0]);
    acc[0][1] = WMMA_BF16(a0, b1, acc[0][1]);
    acc[0][2] = WMMA_BF16(a0, b2, acc[0][2]);
    acc[0][3] = WMMA_BF16(a0, b3, acc[0][3]);
    acc[1][0] = WMMA_BF16(a1, b0, acc[1][0]);
    acc[1][1] = WMMA_BF16(a1, b1, acc[1][1]);
    acc[1][2] = WMMA_BF16(a1, b2, acc[1][2]);
    acc[1][3] = WMMA_BF16(a1, b3, acc[1][3]);
  }

  if (bias) {
#pragma unroll
    for (int t = 0; t < 4; ++t) {
      float bb = bias[n0 + t * 16 + lr];
#pragma unroll
      for (int mi = 0; mi < 2; ++mi)
#pragma unroll
        for (int j = 0; j < 8; ++j) acc[mi][t][j] += bb;
    }
  }

  if (mode == 0) {
    float* O = (float*)out;
#pragma unroll
    for (int mi = 0; mi < 2; ++mi)
#pragma unroll
      for (int t = 0; t < 4; ++t)
#pragma unroll
        for (int j = 0; j < 8; ++j) {
          int m = m0 + mi * 16 + j + hi * 8;
          O[(size_t)m * N + (n0 + t * 16 + lr)] = acc[mi][t][j];
        }
  } else {  // mode 2: V transpose to [B, NKV, HD, S] bf16
    bf16_t* O = (bf16_t*)out;
#pragma unroll
    for (int mi = 0; mi < 2; ++mi)
#pragma unroll
      for (int t = 0; t < 4; ++t)
#pragma unroll
        for (int j = 0; j < 8; ++j) {
          int m = m0 + mi * 16 + j + hi * 8;
          int n = n0 + t * 16 + lr;
          int b = m / S_, s = m - b * S_;
          int kv = n >> 6, d = n & 63;
          O[(((size_t)b * NKV_ + kv) * HD_ + d) * S_ + s] = (bf16_t)acc[mi][t][j];
        }
  }
}

// --------------------------------------------------------------------- RoPE
// raw: [B, S, nheads*64] f32 -> dst: [B, nheads, S, 64] bf16 (scaled).
__global__ void rope_kernel(const float* __restrict__ raw,
                            const float* __restrict__ cost,
                            const float* __restrict__ sint,
                            bf16_t* __restrict__ dst,
                            int nheads, float scale, int total) {
  int idx = blockIdx.x * blockDim.x + threadIdx.x;
  if (idx >= total) return;
  int p = idx & 31;
  int t = idx >> 5;
  int h = t % nheads; t /= nheads;
  int s = t % S_;
  int b = t / S_;
  size_t rbase = ((size_t)(b * S_ + s) * nheads + h) * HD_;
  float x1 = raw[rbase + p];
  float x2 = raw[rbase + p + 32];
  size_t cbase = (size_t)(b * S_ + s) * HD_;
  float c1 = cost[cbase + p],      s1 = sint[cbase + p];
  float c2 = cost[cbase + p + 32], s2 = sint[cbase + p + 32];
  float o1 = (x1 * c1 - x2 * s1) * scale;
  float o2 = (x2 * c2 + x1 * s2) * scale;
  size_t dbase = (((size_t)b * nheads + h) * S_ + s) * HD_;
  dst[dbase + p]      = (bf16_t)o1;
  dst[dbase + p + 32] = (bf16_t)o2;
}

// ---------------------------------------------------- causal flash attention
// Q: [B,NH,S,64] bf16 (pre-scaled by 1/sqrt(64)), Kr: [B,NKV,S,64] bf16,
// Vt: [B,NKV,64,S] bf16, Out: [B,S,NH*64] bf16.
// One wave per 16-query tile; 64-key blocks (16 WMMAs per softmax round).
__global__ void flash_attn_kernel(const bf16_t* __restrict__ Q,
                                  const bf16_t* __restrict__ Kr,
                                  const bf16_t* __restrict__ Vt,
                                  bf16_t* __restrict__ Out) {
  __shared__ __align__(16) bf16_t lds_p[8][16][64];  // per-wave P tile (16KB)

  const int lane = threadIdx.x & 31;
  const int wave = threadIdx.x >> 5;
  const int lr   = lane & 15;
  const int hi   = lane >> 4;
  const int d0   = hi * 8;

  const int b  = blockIdx.z;
  const int h  = blockIdx.y;
  const int kv = h / (NH_ / NKV_);
  const int qt = blockIdx.x * 8 + wave;   // wave-uniform query tile

  const bf16_t* Qb = Q  + (((size_t)b * NH_  + h)  * S_) * HD_;
  const bf16_t* Kb = Kr + (((size_t)b * NKV_ + kv) * S_) * HD_;
  const bf16_t* Vb = Vt + (((size_t)b * NKV_ + kv) * HD_) * S_;

  // Q A-fragments for head-dim chunks [0..31] and [32..63]
  const bf16_t* qrow = Qb + (size_t)(qt * 16 + lr) * HD_;
  v16bf aq0 = load_frag16(qrow + d0,      qrow + d0 + 16);
  v16bf aq1 = load_frag16(qrow + 32 + d0, qrow + 32 + d0 + 16);

  float mrow[8], lrow[8];
#pragma unroll
  for (int j = 0; j < 8; ++j) { mrow[j] = -3.0e38f; lrow[j] = 0.0f; }
  v8f oacc[4] = {};

  const int kend = qt * 16 + 16;
  for (int kb = 0; kb < kend; kb += 64) {      // wave-uniform trip count
    // ---- scores: 16 queries x 64 keys (four 16x16 tiles)
    v8f s[4];
#pragma unroll
    for (int c = 0; c < 4; ++c) {
      const bf16_t* krow = Kb + (size_t)(kb + c * 16 + lr) * HD_;
      v16bf kf0 = load_frag16(krow + d0,      krow + d0 + 16);
      v16bf kf1 = load_frag16(krow + 32 + d0, krow + 32 + d0 + 16);
      v8f sc = {};
      sc = WMMA_BF16(aq0, kf0, sc);
      sc = WMMA_BF16(aq1, kf1, sc);
      s[c] = sc;
    }

    const bool needmask = (kb + 63) > (qt * 16);  // wave-uniform
    float cfac[8];
#pragma unroll
    for (int j = 0; j < 8; ++j) {
      float v0 = s[0][j], v1 = s[1][j], v2 = s[2][j], v3 = s[3][j];
      if (needmask) {
        int qi = qt * 16 + j + hi * 8;
        v0 = (kb + lr      > qi) ? -3.0e38f : v0;   // v_cndmask, EXEC untouched
        v1 = (kb + 16 + lr > qi) ? -3.0e38f : v1;
        v2 = (kb + 32 + lr > qi) ? -3.0e38f : v2;
        v3 = (kb + 48 + lr > qi) ? -3.0e38f : v3;
      }
      // row max across 64 columns: local 4-way tree, then 16-lane butterfly
      float mloc = fmaxf(fmaxf(v0, v1), fmaxf(v2, v3));
      mloc = fmaxf(mloc, __shfl_xor(mloc, 1));
      mloc = fmaxf(mloc, __shfl_xor(mloc, 2));
      mloc = fmaxf(mloc, __shfl_xor(mloc, 4));
      mloc = fmaxf(mloc, __shfl_xor(mloc, 8));
      float mn   = fmaxf(mrow[j], mloc);
      float corr = __expf(mrow[j] - mn);
      mrow[j] = mn;
      float p0 = __expf(v0 - mn);
      float p1 = __expf(v1 - mn);
      float p2 = __expf(v2 - mn);
      float p3 = __expf(v3 - mn);
      lrow[j] = lrow[j] * corr + ((p0 + p1) + (p2 + p3));
      cfac[j] = corr;
      int r = j + hi * 8;
      lds_p[wave][r][lr]      = (bf16_t)p0;
      lds_p[wave][r][16 + lr] = (bf16_t)p1;
      lds_p[wave][r][32 + lr] = (bf16_t)p2;
      lds_p[wave][r][48 + lr] = (bf16_t)p3;
    }
#pragma unroll
    for (int t = 0; t < 4; ++t)
#pragma unroll
      for (int j = 0; j < 8; ++j) oacc[t][j] *= cfac[j];

    // P in A-layout from LDS (same-wave DS ops are in-order; no barrier needed)
    const bf16_t* prow = &lds_p[wave][lr][0];
    v16bf pf0 = load_frag16(prow + d0,      prow + d0 + 16);       // keys 0..31
    v16bf pf1 = load_frag16(prow + 32 + d0, prow + 32 + d0 + 16);  // keys 32..63

    // ---- P x V: 4 head-dim tiles of 16, two key chunks each
#pragma unroll
    for (int t = 0; t < 4; ++t) {
      const bf16_t* vr = Vb + (size_t)(t * 16 + lr) * S_;
      v16bf vf0 = load_frag16(vr + kb + d0,      vr + kb + d0 + 16);
      v16bf vf1 = load_frag16(vr + kb + 32 + d0, vr + kb + 32 + d0 + 16);
      oacc[t] = WMMA_BF16(pf0, vf0, oacc[t]);
      oacc[t] = WMMA_BF16(pf1, vf1, oacc[t]);
    }
  }

  // final row sums and normalized store
  float linv[8];
#pragma unroll
  for (int j = 0; j < 8; ++j) {
    float lv = lrow[j];
    lv += __shfl_xor(lv, 1);
    lv += __shfl_xor(lv, 2);
    lv += __shfl_xor(lv, 4);
    lv += __shfl_xor(lv, 8);
    linv[j] = 1.0f / lv;
  }
#pragma unroll
  for (int j = 0; j < 8; ++j) {
    int q = qt * 16 + j + hi * 8;
    size_t base = (size_t)(b * S_ + q) * (NH_ * HD_) + (size_t)h * HD_;
    Out[base + 0 * 16 + lr] = (bf16_t)(oacc[0][j] * linv[j]);
    Out[base + 1 * 16 + lr] = (bf16_t)(oacc[1][j] * linv[j]);
    Out[base + 2 * 16 + lr] = (bf16_t)(oacc[2][j] * linv[j]);
    Out[base + 3 * 16 + lr] = (bf16_t)(oacc[3][j] * linv[j]);
  }
}

// ----------------------------------------------------------------- launcher
extern "C" void kernel_launch(void* const* d_in, const int* in_sizes, int n_in,
                              void* d_out, int out_size, void* d_ws, size_t ws_size,
                              hipStream_t stream) {
  (void)in_sizes; (void)n_in; (void)out_size; (void)ws_size;

  const float* hs   = (const float*)d_in[0];
  const float* cosr = (const float*)d_in[1];
  const float* sinr = (const float*)d_in[2];
  const float* wq   = (const float*)d_in[3];
  const float* bq   = (const float*)d_in[4];
  const float* wk   = (const float*)d_in[5];
  const float* bk   = (const float*)d_in[6];
  const float* wv   = (const float*)d_in[7];
  const float* bv   = (const float*)d_in[8];
  const float* wo   = (const float*)d_in[9];
  float* outp = (float*)d_out;

  char* ws = (char*)d_ws;
  size_t off = 0;
  auto walloc = [&](size_t bytes) -> void* {
    void* p = ws + off;
    off = (off + bytes + 255) & ~(size_t)255;
    return p;
  };

  const size_t nHS  = (size_t)B_ * S_ * H_;          // 3,670,016
  const size_t nWQ  = (size_t)NH_ * HD_ * H_;        //   802,816
  const size_t nWKV = (size_t)NKV_ * HD_ * H_;       //   114,688
  const size_t nKV  = (size_t)B_ * S_ * NKV_ * HD_;  //   524,288

  bf16_t* hs_bf = (bf16_t*)walloc(nHS * 2);
  bf16_t* wq_bf = (bf16_t*)walloc(nWQ * 2);
  bf16_t* wk_bf = (bf16_t*)walloc(nWKV * 2);
  bf16_t* wv_bf = (bf16_t*)walloc(nWKV * 2);
  bf16_t* wo_bf = (bf16_t*)walloc(nWQ * 2);
  float*  q_raw = (float*)walloc(nHS * 4);
  float*  k_raw = (float*)walloc(nKV * 4);
  bf16_t* q_rot = (bf16_t*)walloc(nHS * 2);
  bf16_t* k_rot = (bf16_t*)walloc(nKV * 2);
  bf16_t* v_t   = (bf16_t*)walloc(nKV * 2);
  bf16_t* attn  = (bf16_t*)walloc(nHS * 2);

  auto cvt = [&](const float* s, bf16_t* d, size_t n) {
    int blocks = (int)((n + 255) / 256);
    cvt_bf16_kernel<<<blocks, 256, 0, stream>>>(s, d, (int)n);
  };
  cvt(hs, hs_bf, nHS);
  cvt(wq, wq_bf, nWQ);
  cvt(wk, wk_bf, nWKV);
  cvt(wv, wv_bf, nWKV);
  cvt(wo, wo_bf, nWQ);

  const int M = B_ * S_;  // 4096

  // Q/K/V projections (V written transposed as bf16)
  gemm_wmma_kernel<<<dim3(M / 256, (NH_ * HD_) / 64), 256, 0, stream>>>(
      hs_bf, wq_bf, bq, q_raw, M, NH_ * HD_, H_, 0);
  gemm_wmma_kernel<<<dim3(M / 256, (NKV_ * HD_) / 64), 256, 0, stream>>>(
      hs_bf, wk_bf, bk, k_raw, M, NKV_ * HD_, H_, 0);
  gemm_wmma_kernel<<<dim3(M / 256, (NKV_ * HD_) / 64), 256, 0, stream>>>(
      hs_bf, wv_bf, bv, v_t, M, NKV_ * HD_, H_, 2);

  // RoPE (Q pre-scaled by 1/sqrt(HD))
  {
    int totq = M * NH_ * 32;
    rope_kernel<<<(totq + 255) / 256, 256, 0, stream>>>(
        q_raw, cosr, sinr, q_rot, NH_, 0.125f, totq);
    int totk = M * NKV_ * 32;
    rope_kernel<<<(totk + 255) / 256, 256, 0, stream>>>(
        k_raw, cosr, sinr, k_rot, NKV_, 1.0f, totk);
  }

  // causal flash attention: 8 query-tiles per block, 14 heads, 2 batches
  flash_attn_kernel<<<dim3(S_ / 128, NH_, B_), 256, 0, stream>>>(
      q_rot, k_rot, v_t, attn);

  // output projection -> f32 d_out
  gemm_wmma_kernel<<<dim3(M / 256, H_ / 64), 256, 0, stream>>>(
      attn, wo_bf, nullptr, outp, M, H_, NH_ * HD_, 0);
}
